// HistogramLoss_27839978013457
// MI455X (gfx1250) — matrix operand order repaired and checked by reading, not compiled, not run on hardware
//
#include <hip/hip_runtime.h>
#include <hip/hip_bf16.h>

// ---------------- problem constants ----------------
#define F        256      // FEATURES
#define NPIX     16384    // H*W
#define NCLS     4        // classes 1..4 (class 0 excluded)
#define NDIR     10000    // DIRECTIONS
#define NDIR_PAD 10048    // padded to a multiple of DTILE
#define DTILE    64       // directions per block in fp_moments (4 waves x 16)
#define TSLOTS   16448    // NPIX + 4*16 padding slack (total padded pixel slots)
#define NSWEEP   8        // Jacobi sweeps

typedef __attribute__((ext_vector_type(16))) __bf16 v16bf;
typedef __attribute__((ext_vector_type(8)))  __bf16 v8bf;
typedef __attribute__((ext_vector_type(8)))  float  v8f;
typedef __attribute__((ext_vector_type(2)))  float  v2f;

// ---------------- workspace layout (byte offsets, 256B aligned) ----------------
#define WS_INTS   0u          // 64 ints: [0..3]=cnt  [4..7]=padcnt  [8..11]=pixoff
#define WS_MIU    256u        // 4*256 f32
#define WS_EVALS  4352u       // 4*256 f32
#define WS_EVSORT 8448u       // 4*256 f32 (descending, clamped 1e-12)
#define WS_ISTD   12544u      // 4*10000 f32 (1/std)
#define WS_M2     172544u     // 4*10000 f32
#define WS_M4     332544u     // 4*10000 f32
#define WS_IDX    492544u     // 16384 int (gathered pixel indices, segmented)
#define WS_COV    558080u     // 4*65536 f32
#define WS_V      1606656u    // 4*65536 f32 (eigenvectors, columns)
#define WS_VSB    2655232u    // 4*65536 bf16 (sorted evecs^T, row-major [eig][feat])
#define WS_PMB    3179520u    // NDIR_PAD*256 bf16 (normalized proj_mat, padded rows zeroed)
#define WS_T      8324096u    // TSLOTS*256 bf16, PIXEL-MAJOR: T[slot][eig]
// total ~16.0 MB

// ========== WMMA bf16 fragment loaders (ISA 7.12.2), vectorized ==========
// A (16x32, 16-bit): lane L: row m=L%16, kh=(L/16)*8; elements are the two
// contiguous runs [k0+kh .. +7] and [k0+kh+16 .. +7] -> two b128 loads.
__device__ inline v16bf hl_load_a_pair(const __bf16* rowbase) {
  v8bf l0 = *(const v8bf*)(rowbase);
  v8bf l1 = *(const v8bf*)(rowbase + 16);
  return __builtin_shufflevector(l0, l1, 0,1,2,3,4,5,6,7,8,9,10,11,12,13,14,15);
}

// ================= K0: per-class pixel gather =================
__global__ void hl_classify(const int* __restrict__ label, int* wsi, int* idx) {
  __shared__ int scnt[NCLS], soff[NCLS], scur[NCLS];
  int t = threadIdx.x;
  if (t < NCLS) { scnt[t] = 0; scur[t] = 0; }
  __syncthreads();
  for (int p = t; p < NPIX; p += 256) {
    int l = label[p];
    if (l >= 1 && l <= NCLS) atomicAdd(&scnt[l-1], 1);
  }
  __syncthreads();
  if (t == 0) {
    int off = 0;
    for (int c = 0; c < NCLS; ++c) {
      int n  = scnt[c];
      int pn = (n + 15) & ~15;
      wsi[c] = n; wsi[4+c] = pn; wsi[8+c] = off; soff[c] = off;
      off += pn;
    }
  }
  __syncthreads();
  for (int p = t; p < NPIX; p += 256) {
    int l = label[p];
    if (l >= 1 && l <= NCLS) {
      int c = l - 1;
      int s = atomicAdd(&scur[c], 1);
      idx[soff[c] + s] = p;
    }
  }
}

// ================= K1: class means =================
__global__ void hl_miu(const float* __restrict__ feat, const int* wsi,
                       const int* __restrict__ idx, float* miu) {
  int c = blockIdx.x, t = threadIdx.x;
  int n = wsi[c], off = wsi[8+c];
  float inv_n = 1.0f / fmaxf((float)n, 1.0f);
  __shared__ int sp[256];
  float s = 0.f;
  const float* row = feat + (size_t)t * NPIX;
  for (int base = 0; base < n; base += 256) {
    int j = base + t;
    sp[t] = (j < n) ? idx[off + j] : 0;
    __syncthreads();
    int lim = min(256, n - base);
    for (int jj = 0; jj < lim; ++jj) s += row[sp[jj]];
    __syncthreads();
  }
  miu[c*F + t] = s * inv_n;
}

// ================= K2: covariance, streaming all pixels, all 4 classes =======
// grid (16,16), block = 1 wave. Branchless label masks; aligned b64 loads.
// m2mat_c = (feat .* w_c) @ feat^T  -> mask applied to A side only.
__global__ void hl_cov(const float* __restrict__ feat, const int* __restrict__ lbl,
                       const int* wsi, const float* __restrict__ miu, float* cov) {
  int i0 = blockIdx.x * 16, j0 = blockIdx.y * 16;
  int lane = threadIdx.x;
  int hi = lane >> 4, lo = lane & 15;
  v8f acc[NCLS];
#pragma unroll
  for (int c = 0; c < NCLS; ++c) acc[c] = (v8f)0.0f;
  const float* arow = feat + (size_t)(i0 + lo)*NPIX;   // A row m=lo
  const float* brow = feat + (size_t)(j0 + lo)*NPIX;   // B col n=lo
  for (int p0 = 0; p0 < NPIX; p0 += 4) {
#if defined(__HIP_DEVICE_COMPILE__) && __has_builtin(__builtin_amdgcn_wmma_f32_16x16x4_f32)
    int p = p0 + hi*2;                                  // lane's K pair
    v2f av = *(const v2f*)(arow + p);
    v2f bv = *(const v2f*)(brow + p);
    int2 lp = *(const int2*)(lbl + p);
    __builtin_prefetch(arow + p0 + 256, 0, 1);
    __builtin_prefetch(brow + p0 + 256, 0, 1);
#pragma unroll
    for (int c = 0; c < NCLS; ++c) {
      v2f am;
      am.x = (lp.x == c + 1) ? av.x : 0.f;
      am.y = (lp.y == c + 1) ? av.y : 0.f;
      acc[c] = __builtin_amdgcn_wmma_f32_16x16x4_f32(false, am, false, bv, (short)0,
                                                     acc[c], false, false);
    }
#else
    // scalar fallback, identical C-layout semantics (lane owns 8 rows x 1 col)
    for (int k = 0; k < 4; ++k) {
      int p = p0 + k;
      int l = lbl[p];
      float fj = brow[p];
#pragma unroll
      for (int c = 0; c < NCLS; ++c) {
        float m = (l == c + 1) ? fj : 0.f;
#pragma unroll
        for (int r = 0; r < 8; ++r)
          acc[c][r] += feat[(size_t)(i0 + r + hi*8)*NPIX + p] * m;
      }
    }
#endif
  }
#pragma unroll
  for (int c = 0; c < NCLS; ++c) {
    float n_safe = fmaxf((float)wsi[c], 1.0f);
    float inv_n  = 1.0f / n_safe;
    float eps    = fminf(fmaxf(1e-5f / n_safe, 1e-8f), 1e-5f); // 1e-8*1000/n clipped
#pragma unroll
    for (int r = 0; r < 8; ++r) {
      int i = i0 + r + hi*8;
      int j = j0 + lo;
      float v = acc[c][r]*inv_n - miu[c*F+i]*miu[c*F+j] + ((i==j) ? eps : 0.f);
      cov[(size_t)c*F*F + (size_t)i*F + j] = v;
    }
  }
}

// ================= K3: parallel cyclic Jacobi eigensolver =================
__device__ inline void hl_pair(int i, int r, int* pp, int* qq) {
  int p, q;
  if (i == 0) { p = 255; q = r % 255; }
  else { p = (r + i) % 255; q = (r + 255 - i) % 255; }
  if (p < q) { *pp = p; *qq = q; } else { *pp = q; *qq = p; }
}

// grid NCLS, block 256, dynamic LDS = (F*F + 256) floats (257 KB of 320 KB WGP LDS)
__global__ void hl_jacobi(const float* __restrict__ cov, float* V, float* evals) {
  extern __shared__ float jsm[];
  float* A  = jsm;              // 256x256
  float* cs = jsm + F*F;        // 128
  float* sn = cs + 128;         // 128
  int c = blockIdx.x, t = threadIdx.x;
  const float* src = cov + (size_t)c*F*F;
  float* Vg = V + (size_t)c*F*F;
  for (int e = t; e < F*F; e += 256) A[e] = src[e];
  for (int e = t; e < F*F; e += 256) Vg[e] = ((e / F) == (e % F)) ? 1.f : 0.f;
  __syncthreads();
  for (int sweep = 0; sweep < NSWEEP; ++sweep) {
    for (int r = 0; r < 255; ++r) {
      if (t < 128) {
        int p, q; hl_pair(t, r, &p, &q);
        float app = A[p*F + p], aqq = A[q*F + q], apq = A[p*F + q];
        float cc = 1.f, ss = 0.f;
        if (fabsf(apq) > 1e-30f) {
          float tau = (aqq - app) / (2.f * apq);
          float tt  = copysignf(1.f, tau) / (fabsf(tau) + sqrtf(1.f + tau*tau));
          cc = rsqrtf(1.f + tt*tt);
          ss = tt * cc;
        }
        cs[t] = cc; sn[t] = ss;
      }
      __syncthreads();
      int i = t >> 1, jb = (t & 1) * 128;
      int p, q; hl_pair(i, r, &p, &q);
      float cc = cs[i], ss = sn[i];
      for (int j = jb; j < jb + 128; ++j) {           // row rotation
        float up = A[p*F + j], uq = A[q*F + j];
        A[p*F + j] = cc*up - ss*uq;
        A[q*F + j] = ss*up + cc*uq;
      }
      __syncthreads();
      for (int j = jb; j < jb + 128; ++j) {           // column rotation
        float vp = A[j*F + p], vq = A[j*F + q];
        A[j*F + p] = cc*vp - ss*vq;
        A[j*F + q] = ss*vp + cc*vq;
      }
      __syncthreads();
      for (int j = jb; j < jb + 128; ++j) {           // eigenvector rotation
        float vp = Vg[j*F + p], vq = Vg[j*F + q];
        Vg[j*F + p] = cc*vp - ss*vq;
        Vg[j*F + q] = ss*vp + cc*vq;
      }
      __syncthreads();
    }
  }
  evals[c*F + t] = A[t*F + t];
}

// ================= K3b: sort descending + pack sorted evecs^T as bf16 ========
__global__ void hl_sort(const float* evals, const float* V,
                        float* evsort, __bf16* vsb) {
  int c = blockIdx.x, t = threadIdx.x;
  __shared__ float se[F];
  se[t] = evals[c*F + t];
  __syncthreads();
  float e = se[t];
  int rank = 0;
  for (int j = 0; j < F; ++j)
    rank += (se[j] > e) || (se[j] == e && j < t);
  evsort[c*F + rank] = fmaxf(e, 1e-12f);
  const float* Vg = V + (size_t)c*F*F;
  __bf16* dst = vsb + (size_t)c*F*F + (size_t)rank*F;
  for (int i = 0; i < F; ++i) dst[i] = (__bf16)Vg[i*F + t];
}

// ================= K4: std per direction + bf16-normalized proj_mat ==========
__global__ void hl_std_pm(const float* __restrict__ pm, const float* evsort,
                          float* istd, __bf16* pmbf) {
  int c = blockIdx.y;
  int d = blockIdx.x * 256 + threadIdx.x;
  if (d >= NDIR) {
    if (c == 0 && d < NDIR_PAD) {          // zero pm pad rows (read by fp_moments)
      for (int k = 0; k < F; ++k) pmbf[(size_t)d*F + k] = (__bf16)0.f;
    }
    return;
  }
  const float* row = pm + (size_t)d * F;
  float n2 = 0.f;
  for (int k = 0; k < F; ++k) { float v = row[k]; n2 += v*v; }
  float inv = rsqrtf(fmaxf(n2, 1e-30f));
  float s = 0.f;
  for (int k = 0; k < F; ++k) {
    float v = row[k] * inv;
    s += v*v*evsort[c*F + k];
    if (c == 0) pmbf[(size_t)d*F + k] = (__bf16)v;
  }
  istd[c*NDIR + d] = rsqrtf(fmaxf(s, 1e-30f));
}

// ============ K5: T = evecs^T @ centered (bf16 WMMA), T pixel-major ==========
// grid (F/16, NPIX/16, NCLS), block = 1 wave
__global__ void hl_project_t(const float* __restrict__ feat, const int* wsi,
                             const int* __restrict__ idx, const float* __restrict__ miu,
                             const __bf16* __restrict__ vsb, __bf16* __restrict__ T) {
  int c = blockIdx.z;
  int pad = wsi[4+c];
  int j0 = blockIdx.y * 16;
  if (j0 >= pad) return;                 // uniform per block
  int n = wsi[c], off = wsi[8+c];
  int m0 = blockIdx.x * 16;
  int lane = threadIdx.x;
  int lo = lane & 15, hi = lane >> 4;
  __shared__ int pix[16];
  __shared__ __attribute__((aligned(32))) __bf16 Bs[16*32];  // [pixel][feat-chunk]
  if (lane < 16) pix[lane] = (j0 + lane < n) ? idx[off + j0 + lane] : -1;
  __syncthreads();
  v8f acc = (v8f)0.0f;
  const __bf16* Ab = vsb + (size_t)c*F*F + (size_t)(m0 + lo)*F + hi*8; // row m=lo, kh=hi*8
  int kb = hi << 3;                                                    // B: kb = hi*8
  for (int k0 = 0; k0 < F; k0 += 32) {
    int f = k0 + lane;
    float mu = miu[c*F + f];
    const float* frow = feat + (size_t)f * NPIX;
#pragma unroll
    for (int j = 0; j < 16; ++j) {
      int p  = pix[j];
      int pc = max(p, 0);
      float v = frow[pc] - mu;             // always-safe load, then select
      Bs[j*32 + lane] = (__bf16)((p >= 0) ? v : 0.f);
    }
    __syncthreads();
    v16bf a = hl_load_a_pair(Ab + k0);
    v16bf b = hl_load_a_pair(&Bs[lo*32 + kb]);
    acc = __builtin_amdgcn_wmma_f32_16x16x32_bf16(false, a, false, b, (short)0,
                                                  acc, false, false);
    __syncthreads();
  }
  // lane writes 8 contiguous eig entries of pixel row (off+j0+lo): one b128 store
  v8bf outv;
#pragma unroll
  for (int r = 0; r < 8; ++r) outv[r] = (__bf16)acc[r];
  *(v8bf*)(T + (size_t)(off + j0 + lo)*F + m0 + hi*8) = outv;
}

// ========== K6: fp = pm @ T fused with m2/m4 accumulation (bf16 WMMA) ========
// grid (ceil(NDIR/64), NCLS), block = 4 waves (128). Each wave owns a 16-dir
// subtile with its pm A-fragments hoisted in registers; the 8 KB contiguous
// T pixel-tile is staged to LDS once and shared by all 4 waves (4x less L2
// traffic on the dominant GEMM). fp never touches memory.
__global__ void hl_fp_moments(const __bf16* __restrict__ pmbf, const __bf16* __restrict__ T,
                              const int* wsi, const float* __restrict__ istd,
                              float* m2o, float* m4o) {
  int c   = blockIdx.y;
  int tid = threadIdx.x;
  int lane = tid & 31;
  int d0  = blockIdx.x * DTILE + (tid >> 5) * 16;      // this wave's 16 directions
  int n = wsi[c], pad = wsi[4+c], off = wsi[8+c];
  float inv_n = 1.0f / fmaxf((float)n, 1.0f);
  int lo = lane & 15, hi = lane >> 4;
  int kb = hi << 3;
  __shared__ __attribute__((aligned(32))) __bf16 Ts[16 * F];  // shared B tile (8 KB)
  float is[8], s2[8], s4[8];
#pragma unroll
  for (int r = 0; r < 8; ++r) {
    int d = d0 + r + hi*8;
    is[r] = istd[c*NDIR + min(d, NDIR - 1)];           // clamped; pad dirs discarded
    s2[r] = 0.f; s4[r] = 0.f;
  }
  v16bf afrag[F/32];                                   // hoisted pm fragments (64 VGPRs)
  const __bf16* Arow = pmbf + (size_t)(d0 + lo)*F + hi*8;
#pragma unroll
  for (int kc = 0; kc < F/32; ++kc) afrag[kc] = hl_load_a_pair(Arow + kc*32);

  for (int p0 = 0; p0 < pad; p0 += 16) {
    // stage 16 contiguous pixel rows (8 KB) cooperatively: 512 x b128 chunks
    const v8bf* src = (const v8bf*)(T + (size_t)(off + p0)*F);
#pragma unroll
    for (int ch = 0; ch < 4; ++ch)
      ((v8bf*)Ts)[tid + ch*128] = src[tid + ch*128];
    __syncthreads();
    v8f acc = (v8f)0.0f;
#pragma unroll
    for (int kc = 0; kc < F/32; ++kc) {
      v16bf b = hl_load_a_pair(&Ts[lo*F + kc*32 + kb]);
      acc = __builtin_amdgcn_wmma_f32_16x16x32_bf16(false, afrag[kc], false, b,
                                                    (short)0, acc, false, false);
    }
#pragma unroll
    for (int r = 0; r < 8; ++r) {
      float v  = acc[r] * is[r];
      float v2 = v*v;
      s2[r] += v2; s4[r] += v2*v2;
    }
    __syncthreads();
  }
#pragma unroll
  for (int r = 0; r < 8; ++r) {
    float a2 = s2[r], a4 = s4[r];
    for (int m = 1; m < 16; m <<= 1) {
      a2 += __shfl_xor(a2, m, 32);
      a4 += __shfl_xor(a4, m, 32);
    }
    int d = d0 + r + hi*8;
    if (lo == 0 && d < NDIR) {
      m2o[c*NDIR + d] = a2 * inv_n;
      m4o[c*NDIR + d] = a4 * inv_n;
    }
  }
}

// ================= K7: final scalar loss =================
__global__ void hl_loss(const int* wsi, const float* m2, const float* m4, float* out) {
  __shared__ float red[256];
  int t = threadIdx.x;
  float tot = 0.f;
  for (int c = 0; c < NCLS; ++c) {
    if (wsi[c] <= 0) continue;
    float part = 0.f;
    for (int d = t; d < NDIR; d += 256) {
      float a = m2[c*NDIR + d], b = m4[c*NDIR + d];
      float e2 = a - 1.f;
      float kk = b - 3.f*a*a;
      part += e2*e2 + kk*kk;
    }
    tot += part * (1.0f / (float)NDIR);
  }
  red[t] = tot;
  __syncthreads();
  for (int s = 128; s > 0; s >>= 1) {
    if (t < s) red[t] += red[t + s];
    __syncthreads();
  }
  if (t == 0) {
    int an = 0;
    for (int c = 0; c < NCLS; ++c) an += (wsi[c] > 0);
    out[0] = red[0] / fmaxf((float)an, 1.0f);   // LOSS_WEIGHT == 1
  }
}

// ================= host launcher =================
extern "C" void kernel_launch(void* const* d_in, const int* in_sizes, int n_in,
                              void* d_out, int out_size, void* d_ws, size_t ws_size,
                              hipStream_t stream) {
  (void)in_sizes; (void)n_in; (void)out_size; (void)ws_size;
  const float* feat = (const float*)d_in[0];   // (1,256,128,128) f32
  const int*   lab  = (const int*)d_in[1];     // (1,1,128,128) i32
  const float* pm   = (const float*)d_in[2];   // (10000,256) f32

  char* ws = (char*)d_ws;
  int*    wsi   = (int*)  (ws + WS_INTS);
  float*  miu   = (float*)(ws + WS_MIU);
  float*  evals = (float*)(ws + WS_EVALS);
  float*  evs   = (float*)(ws + WS_EVSORT);
  float*  istd  = (float*)(ws + WS_ISTD);
  float*  m2    = (float*)(ws + WS_M2);
  float*  m4    = (float*)(ws + WS_M4);
  int*    idx   = (int*)  (ws + WS_IDX);
  float*  cov   = (float*)(ws + WS_COV);
  float*  V     = (float*)(ws + WS_V);
  __bf16* vsb   = (__bf16*)(ws + WS_VSB);
  __bf16* pmbf  = (__bf16*)(ws + WS_PMB);
  __bf16* T     = (__bf16*)(ws + WS_T);
  float*  out   = (float*)d_out;

  hl_classify<<<1, 256, 0, stream>>>(lab, wsi, idx);
  hl_miu<<<NCLS, 256, 0, stream>>>(feat, wsi, idx, miu);
  hl_cov<<<dim3(16, 16), 32, 0, stream>>>(feat, lab, wsi, miu, cov);
  hl_jacobi<<<NCLS, 256, (F*F + 256) * sizeof(float), stream>>>(cov, V, evals);
  hl_sort<<<NCLS, 256, 0, stream>>>(evals, V, evs, vsb);
  hl_std_pm<<<dim3((NDIR_PAD + 255) / 256, NCLS), 256, 0, stream>>>(pm, evs, istd, pmbf);
  hl_project_t<<<dim3(F / 16, NPIX / 16, NCLS), 32, 0, stream>>>(feat, wsi, idx, miu, vsb, T);
  hl_fp_moments<<<dim3((NDIR + DTILE - 1) / DTILE, NCLS), 128, 0, stream>>>(pmbf, T, wsi, istd, m2, m4);
  hl_loss<<<1, 256, 0, stream>>>(wsi, m2, m4, out);
}
